// BagModel_6803228197419
// MI455X (gfx1250) — matrix operand order repaired and checked by reading, not compiled, not run on hardware
//
#include <hip/hip_runtime.h>

#define N_ROWS 262144
#define D_IN   1024
#define D_H    512
#define N_BAGS 512
#define N_CLS  2
#define TM     64
#define KSTEP  32
#define NSTEPS (D_IN / KSTEP)
#define SLAB_ELEMS (D_H * KSTEP)        // 16384 bf16 = 32 KB

typedef __attribute__((ext_vector_type(16))) __bf16 v16bf;
typedef __attribute__((ext_vector_type(8)))  __bf16 v8bf;
typedef __attribute__((ext_vector_type(8)))  float  v8f;
typedef __attribute__((ext_vector_type(4)))  int    v4i;

#define AS1 __attribute__((address_space(1)))
#define AS3 __attribute__((address_space(3)))

#if __has_builtin(__builtin_amdgcn_global_load_async_to_lds_b128)
#define HAVE_ASYNC_LDS 1
#else
#define HAVE_ASYNC_LDS 0
#endif

__device__ __forceinline__ void wait_async0() {
#if __has_builtin(__builtin_amdgcn_s_wait_asynccnt)
    __builtin_amdgcn_s_wait_asynccnt(0);
#else
    asm volatile("s_wait_asynccnt 0x0" ::: "memory");
#endif
}

// Copy one pre-swizzled 32KB W1 slab (global bf16) into an LDS buffer.
// 512 threads x 64B each, via gfx1250 async-to-LDS (ASYNCcnt tracked).
__device__ __forceinline__ void stage_slab(const __bf16* __restrict__ W1t,
                                           __bf16* dstLds, int slab, int t) {
    const char* gsrc = (const char*)(W1t + (size_t)slab * SLAB_ELEMS) + t * 64;
#if HAVE_ASYNC_LDS
    AS3 char* l = (AS3 char*)dstLds + t * 64;
    #pragma unroll
    for (int j = 0; j < 4; ++j)
        __builtin_amdgcn_global_load_async_to_lds_b128(
            (AS1 v4i*)(gsrc + j * 16), (AS3 v4i*)(l + j * 16), 0, 0);
#else
    char* l = (char*)dstLds + t * 64;
    #pragma unroll
    for (int j = 0; j < 4; ++j) {
        uint4 v = *(const uint4*)(gsrc + j * 16);
        *(uint4*)(l + j * 16) = v;
    }
#endif
}

// One-time: W1 [K,D_H] f32 -> bf16, swizzled into B-fragment order:
// W1t[kslab][col][k], k = 0..31 contiguous (exactly the LDS layout WMMA wants).
__global__ __launch_bounds__(256) void w1_prep_kernel(
    const float* __restrict__ W1, __bf16* __restrict__ W1t)
{
    int idx = blockIdx.x * 256 + threadIdx.x;   // 0..16383 = (kb, c)
    int kb = idx >> 9;                          // 0..31
    int c  = idx & 511;
    v8bf tv[4];
    #pragma unroll
    for (int j = 0; j < 4; ++j)
        #pragma unroll
        for (int k = 0; k < 8; ++k)
            tv[j][k] = (__bf16)W1[(size_t)(kb * KSTEP + j * 8 + k) * D_H + c];
    __bf16* dst = W1t + ((size_t)kb * D_H + c) * KSTEP;   // 64B aligned
    #pragma unroll
    for (int j = 0; j < 4; ++j)
        *reinterpret_cast<v8bf*>(dst + j * 8) = tv[j];
}

// Fused: h = relu(x@W1 + b1), segment-summed (sorted ids) into bagSums.
// Grid: N_ROWS/64 blocks, 512 threads (16 waves = 4 row-groups x 4 col-groups).
// Wave = 16 rows x 128 cols = 8 x v_wmma_f32_16x16x32_bf16 accumulators.
// Pipelining: W slab double-buffered in LDS via async-to-LDS; x (A-fragment)
// double-buffered in registers one K-step ahead.
__global__ __launch_bounds__(512) void gemm_bag_kernel(
    const float* __restrict__ x, const int* __restrict__ ids,
    const __bf16* __restrict__ W1t, const float* __restrict__ b1,
    float* __restrict__ bagSums)
{
    __shared__ __align__(128) __bf16 Wt[2][SLAB_ELEMS];   // double-buffered, 64 KB

    const int t       = threadIdx.x;
    const int lane    = t & 31;
    const int wave    = t >> 5;
    const int waveRow = wave & 3;       // 0..3 -> 16-row group
    const int waveCol = wave >> 2;      // 0..3 -> 128-col group
    const int rowBase = blockIdx.x * TM;

    const int halfHi = (lane >> 4) & 1;   // 0: lanes 0-15, 1: lanes 16-31
    const int laneN  = lane & 15;
    const int myRow  = rowBase + waveRow * 16 + laneN;
    // A k order used here (matches B): lanes 0-15 hold K=0..15; lanes 16-31 hold K=16..31
    const int oA     = halfHi ? 4 : 0;          // float4 offset inside 32-wide K slab
    const int koffb  = halfHi ? 16 : 0;         // B: low lanes K=0..15, high lanes K=16..31

    v8f acc[8];
    #pragma unroll
    for (int i = 0; i < 8; ++i) { v8f z = {}; acc[i] = z; }

    const float4* xr = reinterpret_cast<const float4*>(x + (size_t)myRow * D_IN);

    // prologue: slab 0 async copy + first A fragment loads
    stage_slab(W1t, &Wt[0][0], 0, t);
    float4 a0 = xr[oA + 0], a1 = xr[oA + 1], a2 = xr[oA + 2], a3 = xr[oA + 3];
#if HAVE_ASYNC_LDS
    wait_async0();
#endif
    __syncthreads();

    for (int i = 0; i < NSTEPS; ++i) {
        const int buf = i & 1;

        // prefetch next step's x fragment (consumed next iteration)
        float4 p0 = a0, p1 = a1, p2 = a2, p3 = a3;
        if (i + 1 < NSTEPS) {
            const float4* xk = xr + (i + 1) * 8;
            p0 = xk[oA + 0]; p1 = xk[oA + 1]; p2 = xk[oA + 2]; p3 = xk[oA + 3];
            // overlap: async-copy next W slab into the other buffer
            stage_slab(W1t, &Wt[buf ^ 1][0], i + 1, t);
        }

        v16bf A;
        A[0]  = (__bf16)a0.x; A[1]  = (__bf16)a0.y; A[2]  = (__bf16)a0.z; A[3]  = (__bf16)a0.w;
        A[4]  = (__bf16)a1.x; A[5]  = (__bf16)a1.y; A[6]  = (__bf16)a1.z; A[7]  = (__bf16)a1.w;
        A[8]  = (__bf16)a2.x; A[9]  = (__bf16)a2.y; A[10] = (__bf16)a2.z; A[11] = (__bf16)a2.w;
        A[12] = (__bf16)a3.x; A[13] = (__bf16)a3.y; A[14] = (__bf16)a3.z; A[15] = (__bf16)a3.w;

        // load all 8 B fragments, then stream the WMMAs back-to-back
        v16bf B[8];
        #pragma unroll
        for (int cb = 0; cb < 8; ++cb) {
            int col = waveCol * 128 + cb * 16 + laneN;
            B[cb] = *reinterpret_cast<const v16bf*>(&Wt[buf][col * KSTEP + koffb]);
        }
        #pragma unroll
        for (int cb = 0; cb < 8; ++cb)
            acc[cb] = __builtin_amdgcn_wmma_f32_16x16x32_bf16(
                false, A, false, B[cb], (short)0, acc[cb], false, false);

        a0 = p0; a1 = p1; a2 = p2; a3 = p3;

#if HAVE_ASYNC_LDS
        wait_async0();
#endif
        __syncthreads();
    }

    // Epilogue: bias + relu, run-length segmented row reduction, atomic flush.
    // C/D layout: VGPR v, lanes 0-15 -> M=v, lanes 16-31 -> M=v+8; N=lane%16
    const int localRowBase = waveRow * 16 + (halfHi ? 8 : 0);
    int bags[8];
    #pragma unroll
    for (int v = 0; v < 8; ++v) bags[v] = ids[rowBase + localRowBase + v];

    #pragma unroll
    for (int cb = 0; cb < 8; ++cb) {
        int n = waveCol * 128 + cb * 16 + laneN;
        float bias = b1[n];
        float run = 0.0f;
        int curBag = bags[0];
        #pragma unroll
        for (int v = 0; v < 8; ++v) {
            float val = fmaxf(acc[cb][v] + bias, 0.0f);
            if (bags[v] != curBag) {
                atomicAdd(&bagSums[(size_t)curBag * D_H + n], run);
                run = 0.0f;
                curBag = bags[v];
            }
            run += val;
        }
        atomicAdd(&bagSums[(size_t)curBag * D_H + n], run);
    }
}

// Per-bag: count via binary search on sorted ids, mean, then [512]x[512,2] dot.
__global__ __launch_bounds__(256) void finalize_kernel(
    const float* __restrict__ bagSums, const int* __restrict__ ids,
    const float* __restrict__ W2, const float* __restrict__ b2,
    float* __restrict__ out)
{
    const int b = blockIdx.x;
    const int t = threadIdx.x;

    auto lower_bound = [&](int key) {
        int lo = 0, hi = N_ROWS;
        while (lo < hi) {
            int mid = (lo + hi) >> 1;
            if (ids[mid] < key) lo = mid + 1; else hi = mid;
        }
        return lo;
    };
    int cnt = lower_bound(b + 1) - lower_bound(b);
    float inv = 1.0f / (float)(cnt > 0 ? cnt : 1);

    float p0 = 0.0f, p1 = 0.0f;
    #pragma unroll
    for (int i = 0; i < 2; ++i) {
        int c = t + i * 256;
        float f = bagSums[(size_t)b * D_H + c] * inv;
        p0 += f * W2[c * N_CLS + 0];
        p1 += f * W2[c * N_CLS + 1];
    }

    __shared__ float r0[256], r1[256];
    r0[t] = p0; r1[t] = p1;
    __syncthreads();
    for (int s = 128; s > 0; s >>= 1) {
        if (t < s) { r0[t] += r0[t + s]; r1[t] += r1[t + s]; }
        __syncthreads();
    }
    if (t == 0) {
        out[b * N_CLS + 0] = r0[0] + b2[0];
        out[b * N_CLS + 1] = r1[0] + b2[1];
    }
}

extern "C" void kernel_launch(void* const* d_in, const int* in_sizes, int n_in,
                              void* d_out, int out_size, void* d_ws, size_t ws_size,
                              hipStream_t stream) {
    const float* x   = (const float*)d_in[0];
    const int*   ids = (const int*)d_in[1];
    const float* W1  = (const float*)d_in[2];
    const float* b1  = (const float*)d_in[3];
    const float* W2  = (const float*)d_in[4];
    const float* b2  = (const float*)d_in[5];
    float* out       = (float*)d_out;

    float*  bagSums = (float*)d_ws;                              // 1 MB
    __bf16* W1t     = (__bf16*)((char*)d_ws + (1u << 20));       // 1 MB, 64B aligned

    (void)hipMemsetAsync(bagSums, 0, (size_t)N_BAGS * D_H * sizeof(float), stream);
    w1_prep_kernel<<<64, 256, 0, stream>>>(W1, W1t);
    gemm_bag_kernel<<<N_ROWS / TM, 512, 0, stream>>>(x, ids, W1t, b1, bagSums);
    finalize_kernel<<<N_BAGS, 256, 0, stream>>>(bagSums, ids, W2, b2, out);
}
